// DiT_37349035606668
// MI455X (gfx1250) — compile-verified, matrix operand order checked by reading, not run on hardware
//
#include <hip/hip_runtime.h>

// ---------------------------------------------------------------------------
// DiT forward for MI455X (gfx1250), wave32 + WMMA bf16 + Tensor Data Mover.
// Compute-bound (~1.3 TFLOP vs ~17us of weight traffic at 23.3 TB/s), so all
// large matmuls go through v_wmma_f32_16x16x32_bf16 with f32 accumulation.
// Weights are converted fp32 -> bf16 and transposed to [N][K] once so both
// WMMA operands load as contiguous 32B-per-lane fragments (k = 16*(lane/16) + s in both).
// A-tiles are staged into double-buffered LDS by the TDM (tensor_load_to_lds,
// TENSORcnt) when the builtin is available, overlapping DMA with WMMA.
// ---------------------------------------------------------------------------

typedef __attribute__((ext_vector_type(16))) __bf16 v16bf;
typedef __attribute__((ext_vector_type(8)))  float  v8f;

#define BATCH  16
#define NTOK   256
#define D_     1024
#define NHEAD  16
#define HD_    64
#define DEPTHL 12
#define FF_    4096
#define MTOT   4096   // BATCH * NTOK

#if defined(__HIP_DEVICE_COMPILE__) && \
    __has_builtin(__builtin_amdgcn_tensor_load_to_lds) && \
    __has_builtin(__builtin_amdgcn_s_wait_tensorcnt)
#define HAVE_TDM 1
#else
#define HAVE_TDM 0
#endif

#if HAVE_TDM
typedef __attribute__((ext_vector_type(4))) unsigned int u32x4;
typedef __attribute__((ext_vector_type(8))) int i32x8;
typedef __attribute__((ext_vector_type(4))) int i32x4;

// Issue one TDM 2D tile load: global (row-major, 2B elements) -> LDS.
//   gsrc        : global address of tile start
//   ldsByteAddr : destination LDS byte address
//   tileK       : tile width in elements  (== tensor_dim0 here; no OOB)
//   tileRows    : tile height in rows     (== tensor_dim1 here)
//   strideElems : row stride in elements  (tensor_dim0_stride)
// D# layout per CDNA5 ISA section 8 (group0 128b, group1 256b).
__device__ __forceinline__ void tdm_load_2d(const void* gsrc, unsigned ldsByteAddr,
                                            unsigned tileK, unsigned tileRows,
                                            unsigned strideElems) {
  unsigned long long ga = (unsigned long long)gsrc;
  u32x4 g0;
  g0[0] = 1u;                                        // count=1, user descriptor
  g0[1] = ldsByteAddr;                               // lds_addr [63:32]
  g0[2] = (unsigned)(ga & 0xffffffffu);              // global_addr lo
  g0[3] = (unsigned)((ga >> 32) & 0x01ffffffu)       // global_addr hi (57b)
          | 0x80000000u;                             // type=2 ("image")
  i32x8 g1;
  g1[0] = 0x00010000;                                // wg_mask=0, data_size=1 (2B)
  g1[1] = (int)(tileK << 16);                        // tensor_dim0[15:0] @ [63:48]
  g1[2] = (int)((tileK >> 16) | (tileRows << 16));   // dim0 hi | tensor_dim1 lo
  g1[3] = (int)((tileRows >> 16) | (tileK << 16));   // dim1 hi | tile_dim0
  g1[4] = (int)(tileRows & 0xffffu);                 // tile_dim1, tile_dim2=0
  g1[5] = (int)strideElems;                          // tensor_dim0_stride lo32
  g1[6] = 0;                                         // stride hi | dim1_stride lo
  g1[7] = 0;
  i32x4 gz4 = {0, 0, 0, 0};
#if defined(__clang_major__) && __clang_major__ >= 23
  i32x8 gz8 = {0, 0, 0, 0, 0, 0, 0, 0};
  __builtin_amdgcn_tensor_load_to_lds(g0, g1, gz4, gz4, gz8, 0);
#else
  __builtin_amdgcn_tensor_load_to_lds(g0, g1, gz4, gz4, 0);
#endif
}
#endif // HAVE_TDM

__device__ __forceinline__ v8f wmma_bf16(v16bf a, v16bf b, v8f c) {
  // D = A(16x32) * B(32x16) + C(16x16 f32)
  return __builtin_amdgcn_wmma_f32_16x16x32_bf16(
      false, a, false, b, (short)0, c, false, false);
}

// --------------------------- weight convert+transpose ----------------------
// O[l][n][k] = (bf16) W[l][k][n]
__global__ void k_convt(const float* __restrict__ W, __bf16* __restrict__ O,
                        int K, int N) {
  size_t per = (size_t)K * N;
  const float* w = W + per * blockIdx.z;
  __bf16* o = O + per * blockIdx.z;
  size_t idx = (size_t)blockIdx.x * blockDim.x + threadIdx.x;
  if (idx >= per) return;
  int k = (int)(idx / N), n = (int)(idx % N);
  o[(size_t)n * K + k] = (__bf16)w[idx];
}

// --------------------------------- patchify --------------------------------
// X0[b*256 + gy*16+gx][c*64 + py*8 + px] = x[b][c][gy*8+py][gx*8+px]
__global__ void k_patchify(const float* __restrict__ x, __bf16* __restrict__ X0) {
  int idx = blockIdx.x * blockDim.x + threadIdx.x;
  if (idx >= MTOT * 192) return;
  int col = idx % 192, row = idx / 192;
  int c = col / 64, rr = col % 64, py = rr >> 3, px = rr & 7;
  int b = row >> 8, n = row & 255, gy = n >> 4, gx = n & 15;
  int iy = gy * 8 + py, ix = gx * 8 + px;
  X0[idx] = (__bf16)x[(((size_t)b * 3 + c) * 128 + iy) * 128 + ix];
}

// ------------------------------ timestep embed ------------------------------
__global__ void k_temb(const float* __restrict__ t, float* __restrict__ te) {
  int idx = blockIdx.x * blockDim.x + threadIdx.x;
  if (idx >= BATCH * 256) return;
  int b = idx >> 8, i = idx & 255;
  int j = i & 127;
  float f = __expf(-9.210340371976184f * (float)j / 128.0f); // ln(1e4)
  float a = t[b] * f;
  te[idx] = (i < 128) ? cosf(a) : sinf(a);
}

// ----------------------- tiny M=16 GEMMs (timestep/adaLN) -------------------
// out[m][n] = sum_k silu?(A[m][k]) * W[k][n] + bias[n]
__global__ void k_small_gemm(const float* __restrict__ A, const float* __restrict__ W,
                             const float* __restrict__ bias, float* __restrict__ out,
                             int K, int N, int siluIn) {
  int idx = blockIdx.x * blockDim.x + threadIdx.x;
  if (idx >= BATCH * N) return;
  int m = idx / N, n = idx % N;
  const float* a = A + (size_t)m * K;
  float s = 0.f;
  for (int k = 0; k < K; ++k) {
    float av = a[k];
    if (siluIn) av = av / (1.0f + __expf(-av));
    s += av * W[(size_t)k * N + n];
  }
  out[idx] = s + bias[n];
}

// --------------------- LayerNorm + modulate (or gamma/beta) -----------------
// one block (256 thr) per row of h[4096][1024]; writes bf16 z
__global__ void k_lnmod(const float* __restrict__ h, const float* __restrict__ mod,
                        const float* __restrict__ gamma, const float* __restrict__ beta,
                        __bf16* __restrict__ z, int useMod) {
  __shared__ float sA[8], sB[8];
  int row = blockIdx.x;
  int tid = threadIdx.x;
  const float* hr = h + (size_t)row * D_;
  float x0 = hr[tid], x1 = hr[tid + 256], x2 = hr[tid + 512], x3 = hr[tid + 768];
  float s = x0 + x1 + x2 + x3;
  float q = x0 * x0 + x1 * x1 + x2 * x2 + x3 * x3;
  for (int m = 1; m < 32; m <<= 1) { s += __shfl_xor(s, m, 32); q += __shfl_xor(q, m, 32); }
  if ((tid & 31) == 0) { sA[tid >> 5] = s; sB[tid >> 5] = q; }
  __syncthreads();
  float ts = 0.f, tq = 0.f;
#pragma unroll
  for (int i = 0; i < 8; ++i) { ts += sA[i]; tq += sB[i]; }
  float mean = ts * (1.0f / 1024.0f);
  float var = tq * (1.0f / 1024.0f) - mean * mean;
  float inv = rsqrtf(var + 1e-5f);
  int b = row >> 8;
  __bf16* zr = z + (size_t)row * D_;
#pragma unroll
  for (int j = 0; j < 4; ++j) {
    int d = tid + j * 256;
    float v = (hr[d] - mean) * inv;
    if (useMod) {
      float sc = mod[(size_t)b * 2048 + d];
      float sh = mod[(size_t)b * 2048 + 1024 + d];
      v = v * (1.f + sc) + sh;
    } else {
      v = v * gamma[d] + beta[d];
    }
    zr[d] = (__bf16)v;
  }
}

// ------------------------------ main WMMA GEMM ------------------------------
// C[M][N] = A[M][K](bf16) * Bt[N][K](bf16)^T + bias + add, optional GELU.
// Block tile 128x64, 8 waves, wave tile 32x32 (2x2 WMMA accumulators).
// A tile (128x64, 16KB) staged into double-buffered LDS by the TDM (wave 0
// issues tensor_load_to_lds; s_wait_tensorcnt 1 keeps next DMA in flight
// behind the current WMMA stage). B fragments stream from global (L2).
// M%128==0, N%64==0, K%64==0 (true for all call sites).
__launch_bounds__(256, 2)
__global__ void k_gemm(const __bf16* __restrict__ A, const __bf16* __restrict__ Bt,
                       const float* __restrict__ bias, const float* addp,
                       float* outF, __bf16* outB,
                       int M, int N, int K, int addMode, int act) {
  __shared__ __align__(128) __bf16 As[2][128 * 64]; // 2 x 16 KB
  const int tid = threadIdx.x;
  const int wid = tid >> 5;
  const int lane = tid & 31;
  const int laneRow = lane & 15;
  const int laneHi = lane >> 4;
  const int bm = blockIdx.y * 128;
  const int bn = blockIdx.x * 64;
  const int wmOff = (wid & 3) * 32;
  const int wn = bn + (wid >> 2) * 32;
  const int nk = K >> 6;

  v8f acc00 = {}, acc01 = {}, acc10 = {}, acc11 = {};

  const __bf16* bptr0 = Bt + (size_t)(wn + laneRow) * K + laneHi * 16;
  const __bf16* bptr1 = bptr0 + (size_t)16 * K;

#if HAVE_TDM
  if (wid == 0) // prologue: DMA first A tile into buffer 0 (LDS offset 0)
    tdm_load_2d(A + (size_t)bm * K, 0u, 64u, 128u, (unsigned)K);
#endif

  for (int kt = 0; kt < nk; ++kt) {
    const int k0 = kt << 6;
    const int buf = kt & 1;
#if HAVE_TDM
    if (wid == 0) {
      if (kt + 1 < nk) {
        // queue next tile into the other buffer, then drain only the older DMA
        tdm_load_2d(A + (size_t)bm * K + (k0 + 64), (unsigned)((buf ^ 1) * 16384),
                    64u, 128u, (unsigned)K);
        __builtin_amdgcn_s_wait_tensorcnt(1);
      } else {
        __builtin_amdgcn_s_wait_tensorcnt(0);
      }
    }
    __syncthreads();   // publish TDM-filled LDS to all waves
#else
    __syncthreads();
    { // fallback: synchronous cooperative copy, 64B per thread
      int idx = tid * 32;
      int r = idx >> 6, c = idx & 63;
      const __bf16* src = A + (size_t)(bm + r) * K + k0 + c;
      *(v16bf*)&As[buf][idx]      = *(const v16bf*)src;
      *(v16bf*)&As[buf][idx + 16] = *(const v16bf*)(src + 16);
    }
    __syncthreads();
#endif
    if (kt + 1 < nk) { // pull next B tiles toward L2 (global_prefetch_b8)
      __builtin_prefetch(bptr0 + k0 + 64, 0, 1);
      __builtin_prefetch(bptr1 + k0 + 64, 0, 1);
    }
    const __bf16* asb = &As[buf][0];
#pragma unroll
    for (int ks = 0; ks < 2; ++ks) {
      v16bf a0 = *(const v16bf*)&asb[(wmOff + laneRow) * 64 + ks * 32 + laneHi * 16];
      v16bf a1 = *(const v16bf*)&asb[(wmOff + 16 + laneRow) * 64 + ks * 32 + laneHi * 16];
      v16bf b0 = *(const v16bf*)(bptr0 + k0 + ks * 32);
      v16bf b1 = *(const v16bf*)(bptr1 + k0 + ks * 32);
      acc00 = wmma_bf16(a0, b0, acc00);
      acc01 = wmma_bf16(a0, b1, acc01);
      acc10 = wmma_bf16(a1, b0, acc10);
      acc11 = wmma_bf16(a1, b1, acc11);
    }
    __syncthreads();   // stage consumed; safe to refill this buffer next round
  }

  const int wm = bm + wmOff;
#pragma unroll
  for (int ti = 0; ti < 2; ++ti) {
#pragma unroll
    for (int tj = 0; tj < 2; ++tj) {
      v8f a = (ti == 0) ? (tj == 0 ? acc00 : acc01) : (tj == 0 ? acc10 : acc11);
      int col = wn + tj * 16 + laneRow;
      float bv = bias ? bias[col] : 0.f;
#pragma unroll
      for (int r = 0; r < 8; ++r) {
        int row = wm + ti * 16 + laneHi * 8 + r;   // C layout: lanes>=16 -> M+8
        float v = a[r] + bv;
        if (addMode == 1)      v += addp[(size_t)row * N + col];          // residual
        else if (addMode == 2) v += addp[(size_t)(row & 255) * N + col];  // pos embed
        if (act == 1) v = 0.5f * v * (1.0f + erff(v * 0.70710678118f));   // exact GELU
        size_t oi = (size_t)row * N + col;
        if (outB) outB[oi] = (__bf16)v;
        else      outF[oi] = v;
      }
    }
  }
}

// ----------------------------------- RoPE -----------------------------------
// qkv[4096][3072] bf16 -> Qr/Kr [bh][n][64] bf16 (rotated), Vt [bh][64][n] bf16
__global__ void k_rope(const __bf16* __restrict__ qkv, __bf16* __restrict__ Qr,
                       __bf16* __restrict__ Kr, __bf16* __restrict__ Vt) {
  int idx = blockIdx.x * blockDim.x + threadIdx.x;
  if (idx >= BATCH * NHEAD * NTOK * 32) return;
  int j = idx & 31;
  int n = (idx >> 5) & 255;
  int bh = idx >> 13;
  int b = bh >> 4, hh = bh & 15;
  int row = b * 256 + n;
  const __bf16* src = qkv + (size_t)row * 3072 + hh * 64;
  int jj = j & 15;
  float base = (j < 16) ? (float)(n >> 4) : (float)(n & 15);   // py then px
  float f = __expf(-0.2878231366242557f * (float)jj);           // ln(1e4)/32
  float ang = base * f;
  float cs = cosf(ang), sn = sinf(ang);
  float q1 = (float)src[j],        q2 = (float)src[32 + j];
  float k1 = (float)src[1024 + j], k2 = (float)src[1024 + 32 + j];
  size_t qo = ((size_t)bh * 256 + n) * 64;
  Qr[qo + j]      = (__bf16)(q1 * cs - q2 * sn);
  Qr[qo + 32 + j] = (__bf16)(q1 * sn + q2 * cs);
  Kr[qo + j]      = (__bf16)(k1 * cs - k2 * sn);
  Kr[qo + 32 + j] = (__bf16)(k1 * sn + k2 * cs);
  Vt[((size_t)bh * 64 + j) * 256 + n]      = src[2048 + j];
  Vt[((size_t)bh * 64 + 32 + j) * 256 + n] = src[2048 + 32 + j];
}

// ------------------------------ fused attention -----------------------------
// One block per (b,h): 512 threads = 16 waves; each wave owns 16 query rows.
// K tile (32KB) DMA'd to LDS by the TDM; S = Q K^T (WMMA), in-register softmax
// (wave32 shfl reductions), O = P V (WMMA) with P staged through per-wave LDS.
__launch_bounds__(512, 1)
__global__ void k_attn(const __bf16* __restrict__ Qr, const __bf16* __restrict__ Kr,
                       const __bf16* __restrict__ Vt, __bf16* __restrict__ O) {
  __shared__ __align__(128) __bf16 Ks[256 * 64];    // 32 KB: K tokens x feats
  __shared__ __align__(64)  __bf16 Ps[16][16 * 32]; // 16 KB: per-wave P staging
  const int bh = blockIdx.x;
  const int b = bh >> 4, hh = bh & 15;
  const int tid = threadIdx.x;
  const int wid = tid >> 5, lane = tid & 31;
  const int laneRow = lane & 15, laneHi = lane >> 4;

#if HAVE_TDM
  if (wid == 0) { // contiguous 32KB: 1-row tile of 16384 2B elements
    tdm_load_2d(Kr + (size_t)bh * 16384, 0u, 16384u, 1u, 16384u);
    __builtin_amdgcn_s_wait_tensorcnt(0);
  }
  __syncthreads();
#else
  { // cooperative K load: 16384 bf16, 32 per thread
    const __bf16* src = Kr + (size_t)bh * 16384 + tid * 32;
    *(v16bf*)&Ks[tid * 32]      = *(const v16bf*)src;
    *(v16bf*)&Ks[tid * 32 + 16] = *(const v16bf*)(src + 16);
  }
  __syncthreads();
#endif

  const int qr0 = wid * 16;
  v16bf aq0, aq1;
  {
    const __bf16* qb = Qr + ((size_t)bh * 256 + qr0 + laneRow) * 64 + laneHi * 16;
    aq0 = *(const v16bf*)qb;
    aq1 = *(const v16bf*)(qb + 32);
  }

  v8f s[16];
  v8f vzero = {};
#pragma unroll
  for (int kt = 0; kt < 16; ++kt) s[kt] = vzero;
#pragma unroll 4
  for (int kt = 0; kt < 16; ++kt) {
    v16bf bk0 = *(const v16bf*)&Ks[(kt * 16 + laneRow) * 64 + laneHi * 16];
    v16bf bk1 = *(const v16bf*)&Ks[(kt * 16 + laneRow) * 64 + 32 + laneHi * 16];
    s[kt] = wmma_bf16(aq0, bk0, s[kt]);
    s[kt] = wmma_bf16(aq1, bk1, s[kt]);
  }

  // softmax over 256 cols; lane holds col kt*16+laneRow, vgpr r -> row r(+8)
  const float scale = 0.125f; // HD^-0.5
  float rmax[8], rsum[8], rinv[8];
#pragma unroll
  for (int r = 0; r < 8; ++r) rmax[r] = -3.0e38f;
  for (int kt = 0; kt < 16; ++kt)
#pragma unroll
    for (int r = 0; r < 8; ++r) rmax[r] = fmaxf(rmax[r], s[kt][r]);
#pragma unroll
  for (int m = 1; m < 16; m <<= 1)
#pragma unroll
    for (int r = 0; r < 8; ++r) rmax[r] = fmaxf(rmax[r], __shfl_xor(rmax[r], m, 32));
#pragma unroll
  for (int r = 0; r < 8; ++r) rsum[r] = 0.f;
  for (int kt = 0; kt < 16; ++kt)
#pragma unroll
    for (int r = 0; r < 8; ++r) {
      float p = __expf((s[kt][r] - rmax[r]) * scale);
      s[kt][r] = p;
      rsum[r] += p;
    }
#pragma unroll
  for (int m = 1; m < 16; m <<= 1)
#pragma unroll
    for (int r = 0; r < 8; ++r) rsum[r] += __shfl_xor(rsum[r], m, 32);
#pragma unroll
  for (int r = 0; r < 8; ++r) rinv[r] = 1.0f / rsum[r];

  // O = P @ V over 8 k-steps of 32 keys
  v8f o0 = vzero, o1 = vzero, o2 = vzero, o3 = vzero;
  __bf16* ps = &Ps[wid][0];
  for (int kc = 0; kc < 8; ++kc) {
    __builtin_amdgcn_wave_barrier();
#pragma unroll
    for (int t = 0; t < 2; ++t) {
      int kt = kc * 2 + t;
#pragma unroll
      for (int r = 0; r < 8; ++r)
        ps[(r + 8 * laneHi) * 32 + t * 16 + laneRow] = (__bf16)(s[kt][r] * rinv[r]);
    }
    __builtin_amdgcn_wave_barrier();
    asm volatile("s_wait_dscnt 0" ::: "memory"); // LDS in-order per wave; fence VGPR returns
    v16bf ap = *(const v16bf*)&ps[laneRow * 32 + laneHi * 16];
    const __bf16* vb = Vt + ((size_t)bh * 64 + laneRow) * 256 + kc * 32 + laneHi * 16;
    o0 = wmma_bf16(ap, *(const v16bf*)(vb),            o0);
    o1 = wmma_bf16(ap, *(const v16bf*)(vb + 16 * 256), o1);
    o2 = wmma_bf16(ap, *(const v16bf*)(vb + 32 * 256), o2);
    o3 = wmma_bf16(ap, *(const v16bf*)(vb + 48 * 256), o3);
    __builtin_amdgcn_wave_barrier();
  }

#pragma unroll
  for (int ot = 0; ot < 4; ++ot) {
    v8f oo = (ot == 0) ? o0 : (ot == 1) ? o1 : (ot == 2) ? o2 : o3;
    int col = hh * 64 + ot * 16 + laneRow;
#pragma unroll
    for (int r = 0; r < 8; ++r) {
      int row = b * 256 + qr0 + laneHi * 8 + r;
      O[(size_t)row * D_ + col] = (__bf16)oo[r];
    }
  }
}

// -------------------------------- unpatchify --------------------------------
// out[b][c][gy*8+py][gx*8+px] = y[b*256+gy*16+gx][py*24+px*3+c]
__global__ void k_unpatch(const float* __restrict__ y, float* __restrict__ out) {
  int idx = blockIdx.x * blockDim.x + threadIdx.x;
  if (idx >= BATCH * 3 * 128 * 128) return;
  int ix = idx & 127;
  int iy = (idx >> 7) & 127;
  int c = (idx >> 14) % 3;
  int b = idx / 49152;
  int gy = iy >> 3, py = iy & 7, gx = ix >> 3, px = ix & 7;
  int row = b * 256 + gy * 16 + gx;
  int col = py * 24 + px * 3 + c;
  out[idx] = y[(size_t)row * 192 + col];
}

// =============================== host driver ================================
extern "C" void kernel_launch(void* const* d_in, const int* in_sizes, int n_in,
                              void* d_out, int out_size, void* d_ws, size_t ws_size,
                              hipStream_t stream) {
  const float* x     = (const float*)d_in[0];
  const float* t     = (const float*)d_in[1];
  const float* Wp    = (const float*)d_in[2];
  const float* bp    = (const float*)d_in[3];
  const float* pos   = (const float*)d_in[4];
  const float* Wt1   = (const float*)d_in[5];
  const float* bt1   = (const float*)d_in[6];
  const float* Wt2   = (const float*)d_in[7];
  const float* bt2   = (const float*)d_in[8];
  const float* Wmod1 = (const float*)d_in[9];
  const float* bmod1 = (const float*)d_in[10];
  const float* Wmod2 = (const float*)d_in[11];
  const float* bmod2 = (const float*)d_in[12];
  const float* Wqkv  = (const float*)d_in[13];
  const float* bqkv  = (const float*)d_in[14];
  const float* Wproj = (const float*)d_in[15];
  const float* bproj = (const float*)d_in[16];
  const float* Wff1  = (const float*)d_in[17];
  const float* bff1  = (const float*)d_in[18];
  const float* Wff2  = (const float*)d_in[19];
  const float* bff2  = (const float*)d_in[20];
  const float* gamma = (const float*)d_in[21];
  const float* beta  = (const float*)d_in[22];
  const float* Wout  = (const float*)d_in[23];
  const float* bout  = (const float*)d_in[24];
  float* out = (float*)d_out;

  char* p = (char*)d_ws;
  auto carve = [&](size_t bytes) -> char* {
    char* r = p;
    p += (bytes + 255) & ~((size_t)255);
    return r;
  };
  // bf16 transposed weights
  __bf16* wWp   = (__bf16*)carve((size_t)1024 * 192 * 2);
  __bf16* wQKV  = (__bf16*)carve((size_t)DEPTHL * 3072 * 1024 * 2);
  __bf16* wProj = (__bf16*)carve((size_t)DEPTHL * 1024 * 1024 * 2);
  __bf16* wFF1  = (__bf16*)carve((size_t)DEPTHL * 4096 * 1024 * 2);
  __bf16* wFF2  = (__bf16*)carve((size_t)DEPTHL * 1024 * 4096 * 2);
  __bf16* wOut  = (__bf16*)carve((size_t)192 * 1024 * 2);
  // activations
  __bf16* X0   = (__bf16*)carve((size_t)MTOT * 192 * 2);
  float*  h    = (float*) carve((size_t)MTOT * D_ * 4);
  __bf16* z    = (__bf16*)carve((size_t)MTOT * D_ * 2);
  __bf16* qkv  = (__bf16*)carve((size_t)MTOT * 3072 * 2);
  __bf16* Qr   = (__bf16*)carve((size_t)256 * 256 * 64 * 2);
  __bf16* Kr   = (__bf16*)carve((size_t)256 * 256 * 64 * 2);
  __bf16* Vt   = (__bf16*)carve((size_t)256 * 64 * 256 * 2);
  __bf16* o    = (__bf16*)carve((size_t)MTOT * D_ * 2);
  __bf16* g    = (__bf16*)carve((size_t)MTOT * FF_ * 2);
  float* temb  = (float*)carve((size_t)BATCH * 256 * 4);
  float* temb1 = (float*)carve((size_t)BATCH * 1024 * 4);
  float* temb2 = (float*)carve((size_t)BATCH * 1024 * 4);
  float* m1    = (float*)carve((size_t)BATCH * 2048 * 4);
  float* m2    = (float*)carve((size_t)BATCH * 2048 * 4);
  float* y     = (float*)carve((size_t)MTOT * 192 * 4);

  dim3 T(256);
  // ---- weight conversion (once per launch; ~1.2 GB traffic, ~50us) ----
  k_convt<<<dim3((192 * 1024 + 255) / 256, 1, 1), T, 0, stream>>>(Wp, wWp, 192, 1024);
  k_convt<<<dim3((1024 * 3072 + 255) / 256, 1, DEPTHL), T, 0, stream>>>(Wqkv, wQKV, 1024, 3072);
  k_convt<<<dim3((1024 * 1024 + 255) / 256, 1, DEPTHL), T, 0, stream>>>(Wproj, wProj, 1024, 1024);
  k_convt<<<dim3((1024 * 4096 + 255) / 256, 1, DEPTHL), T, 0, stream>>>(Wff1, wFF1, 1024, 4096);
  k_convt<<<dim3((4096 * 1024 + 255) / 256, 1, DEPTHL), T, 0, stream>>>(Wff2, wFF2, 4096, 1024);
  k_convt<<<dim3((1024 * 192 + 255) / 256, 1, 1), T, 0, stream>>>(Wout, wOut, 1024, 192);

  // ---- patch embed: h = patchify(x) @ Wp + bp + pos_embed ----
  k_patchify<<<(MTOT * 192 + 255) / 256, T, 0, stream>>>(x, X0);
  k_gemm<<<dim3(1024 / 64, MTOT / 128), T, 0, stream>>>(
      X0, wWp, bp, pos, h, nullptr, MTOT, 1024, 192, /*addMode=*/2, /*act=*/0);

  // ---- timestep conditioning ----
  k_temb<<<(BATCH * 256 + 255) / 256, T, 0, stream>>>(t, temb);
  k_small_gemm<<<(BATCH * 1024 + 255) / 256, T, 0, stream>>>(temb, Wt1, bt1, temb1, 256, 1024, 0);
  k_small_gemm<<<(BATCH * 1024 + 255) / 256, T, 0, stream>>>(temb1, Wt2, bt2, temb2, 1024, 1024, 1);
  // c = silu(temb2) is fused via siluIn=1 into the modulation GEMMs below

  for (int i = 0; i < DEPTHL; ++i) {
    // adaLN 1
    k_small_gemm<<<(BATCH * 2048 + 255) / 256, T, 0, stream>>>(
        temb2, Wmod1 + (size_t)i * 1024 * 2048, bmod1 + i * 2048, m1, 1024, 2048, 1);
    k_lnmod<<<MTOT, T, 0, stream>>>(h, m1, nullptr, nullptr, z, 1);
    // qkv projection (bf16 out)
    k_gemm<<<dim3(3072 / 64, MTOT / 128), T, 0, stream>>>(
        z, wQKV + (size_t)i * 3072 * 1024, bqkv + i * 3072, nullptr,
        nullptr, qkv, MTOT, 3072, 1024, 0, 0);
    // RoPE + V transpose
    k_rope<<<(BATCH * NHEAD * NTOK * 32 + 255) / 256, T, 0, stream>>>(qkv, Qr, Kr, Vt);
    // fused attention per head
    k_attn<<<BATCH * NHEAD, 512, 0, stream>>>(Qr, Kr, Vt, o);
    // output projection + residual into h
    k_gemm<<<dim3(1024 / 64, MTOT / 128), T, 0, stream>>>(
        o, wProj + (size_t)i * 1024 * 1024, bproj + i * 1024, h,
        h, nullptr, MTOT, 1024, 1024, 1, 0);
    // adaLN 2
    k_small_gemm<<<(BATCH * 2048 + 255) / 256, T, 0, stream>>>(
        temb2, Wmod2 + (size_t)i * 1024 * 2048, bmod2 + i * 2048, m2, 1024, 2048, 1);
    k_lnmod<<<MTOT, T, 0, stream>>>(h, m2, nullptr, nullptr, z, 1);
    // MLP: g = gelu(z @ Wff1 + bff1) (bf16) ; h += g @ Wff2 + bff2
    k_gemm<<<dim3(FF_ / 64, MTOT / 128), T, 0, stream>>>(
        z, wFF1 + (size_t)i * 4096 * 1024, bff1 + i * 4096, nullptr,
        nullptr, g, MTOT, FF_, 1024, 0, /*gelu=*/1);
    k_gemm<<<dim3(1024 / 64, MTOT / 128), T, 0, stream>>>(
        g, wFF2 + (size_t)i * 1024 * 4096, bff2 + i * 1024, h,
        h, nullptr, MTOT, 1024, FF_, 1, 0);
  }

  // ---- final LN + output projection + unpatchify ----
  k_lnmod<<<MTOT, T, 0, stream>>>(h, nullptr, gamma, beta, z, 0);
  k_gemm<<<dim3(192 / 64, MTOT / 128), T, 0, stream>>>(
      z, wOut, bout, nullptr, y, nullptr, MTOT, 192, 1024, 0, 0);
  k_unpatch<<<(BATCH * 3 * 128 * 128 + 255) / 256, T, 0, stream>>>(y, out);
}
